// MFN_3822520894008
// MI455X (gfx1250) — matrix operand order repaired and checked by reading, not compiled, not run on hardware
//
#include <hip/hip_runtime.h>

// ===========================================================================
// MFN fused forward for MI455X (gfx1250, wave32, WMMA f32_16x16x32_f16).
//
// Grid = 16 blocks; each owns a 16-row batch tile (M=16 == WMMA M) and runs
// the full 200-step recurrence privately (recurrence is independent per row,
// so no grid-wide sync). Weights pre-converted f32->f16 into d_ws (3.4MB,
// resident in the 192MB L2). State (c, mem) fp32 in LDS; f16 mirrors feed
// WMMA A operands. Dynamic LDS = 153,600 B (< 320KB WGP budget).
//
// This revision software-pipelines all GEMM fragment loads (double-buffered
// SSA values) so the backend can issue next-k loads before waiting on the
// current group (partial s_wait_loadcnt instead of wait-to-zero before every
// v_wmma), and prefetches the next timestep's x rows (global_prefetch_b8).
// ===========================================================================

typedef _Float16 v8h  __attribute__((ext_vector_type(8)));
typedef _Float16 v16h __attribute__((ext_vector_type(16)));
typedef float    v8f  __attribute__((ext_vector_type(8)));

union F16x16 { v16h v; v8h h[2]; };

constexpr int BB  = 256;   // batch
constexpr int TT  = 200;   // timesteps
constexpr int TDI = 768;   // text dim
constexpr int ADI = 512;   // audio dim
constexpr int VDI = 512;   // video dim
constexpr int HH  = 128;   // hidden / MEM

constexpr size_t SMEM_BYTES = 153600;

__device__ __forceinline__ float sigf(float x) { return 1.f / (1.f + __expf(-x)); }

// A fragment (16x32 f16) from row-major LDS, stride lda (halfs).
// lane L: row = L&15, halves {kb..kb+7, kb+16..kb+23}, kb = (L>>4)*8.
__device__ __forceinline__ v16h load_a_frag(const _Float16* A, int lda, int k0, int lane) {
  int row = lane & 15;
  int kb  = (lane >> 4) * 8;
  const _Float16* p = A + row * lda + k0 + kb;
  F16x16 f;
  f.h[0] = *(const v8h*)(p);
  f.h[1] = *(const v8h*)(p + 16);
  return f.v;
}

// B fragment (32x16 f16) for X @ W^T with W row-major [N][K], stride ldb.
// lane L: col = n0 + (L&15), halves {kb..kb+7, kb+16..kb+23}, kb = k0 + (L>>4)*8.
__device__ __forceinline__ v16h load_b_frag(const _Float16* Bm, int ldb, int n0, int k0, int lane) {
  int col = n0 + (lane & 15);
  const _Float16* p = Bm + col * ldb + k0 + (lane >> 4) * 8;
  F16x16 f;
  f.h[0] = *(const v8h*)(p);
  f.h[1] = *(const v8h*)(p + 16);
  return f.v;
}

__device__ __forceinline__ v8f wmma32(v8f acc, v16h a, v16h b) {
  return __builtin_amdgcn_wmma_f32_16x16x32_f16(false, a, false, b, (short)0, acc, false, false);
}

// Single-accumulator GEMM, K >= 64, 2-deep software pipeline: fragments for
// k+1 are loaded into fresh registers before the WMMA consuming k, so the
// scheduler can overlap load latency with matrix math (partial waits).
__device__ __forceinline__ void gemm_pipe(v8f& acc, const _Float16* A, int lda, int K,
                                          const _Float16* Bm, int ldb, int n0, int lane) {
  v16h a0 = load_a_frag(A, lda, 0, lane);
  v16h b0 = load_b_frag(Bm, ldb, n0, 0, lane);
  v16h a1 = load_a_frag(A, lda, 32, lane);
  v16h b1 = load_b_frag(Bm, ldb, n0, 32, lane);
#pragma unroll
  for (int k = 64; k < K; k += 32) {
    acc = wmma32(acc, a0, b0);
    a0 = a1; b0 = b1;
    a1 = load_a_frag(A, lda, k, lane);
    b1 = load_b_frag(Bm, ldb, n0, k, lane);
  }
  acc = wmma32(acc, a0, b0);
  acc = wmma32(acc, a1, b1);
}

// K=128 GEMM: all 4 fragment pairs issued up-front, 4 back-to-back WMMAs.
__device__ __forceinline__ v8f gemm_k128(const _Float16* A, int lda,
                                         const _Float16* Bm, int ldb, int n0, int lane) {
  v16h a0 = load_a_frag(A, lda, 0,  lane), b0 = load_b_frag(Bm, ldb, n0, 0,  lane);
  v16h a1 = load_a_frag(A, lda, 32, lane), b1 = load_b_frag(Bm, ldb, n0, 32, lane);
  v16h a2 = load_a_frag(A, lda, 64, lane), b2 = load_b_frag(Bm, ldb, n0, 64, lane);
  v16h a3 = load_a_frag(A, lda, 96, lane), b3 = load_b_frag(Bm, ldb, n0, 96, lane);
  v8f acc = {};
  acc = wmma32(acc, a0, b0);
  acc = wmma32(acc, a1, b1);
  acc = wmma32(acc, a2, b2);
  acc = wmma32(acc, a3, b3);
  return acc;
}

// D tile (16x16 f32) -> f16 LDS with bias (+ optional relu).
template <int ACT>  // 0 = identity, 1 = relu
__device__ __forceinline__ void epilogue_f16(_Float16* D, int ldd, int n0, v8f acc,
                                             const float* bias, int lane) {
  int n  = n0 + (lane & 15);
  int m0 = (lane >> 4) * 8;
  float b = bias[n];
#pragma unroll
  for (int v = 0; v < 8; ++v) {
    float x = acc[v] + b;
    if (ACT == 1) x = fmaxf(x, 0.f);
    D[(m0 + v) * ldd + n] = (_Float16)x;
  }
}

// Cooperative load of x_t tile [16][K] f32 global -> f16 LDS, with
// global_prefetch of the next timestep's rows (one per 128B line).
__device__ __forceinline__ void load_x_tile(_Float16* dst, const float* src,
                                            int rowbase, int t, int K, int tid) {
  int q = K >> 2;
  for (int idx = tid; idx < 16 * q; idx += 256) {
    int r  = idx / q;
    int c4 = (idx - r * q) << 2;
    const float* gp = src + ((size_t)(rowbase + r) * TT + t) * K + c4;
    float4 x = *(const float4*)gp;
    _Float16* d = dst + r * K + c4;
    d[0] = (_Float16)x.x; d[1] = (_Float16)x.y;
    d[2] = (_Float16)x.z; d[3] = (_Float16)x.w;
    if (((c4 & 31) == 0) && (t + 1 < TT))
      __builtin_prefetch(gp + K, 0, 0);  // same row, next timestep
  }
}

// A fragment + 4 gate B-fragments for one k-tile of the unified [x|h] K loop.
struct Frag5 { v16h a, b0, b1, b2, b3; };

__device__ __forceinline__ Frag5 load_cell_frags(
    int kt, int KxT, const _Float16* sX, int Kx, const _Float16* hA,
    const _Float16* Wih, const _Float16* Whh, int n0, int lane) {
  Frag5 f;
  if (kt < KxT) {
    int k = kt * 32;
    f.a  = load_a_frag(sX, Kx, k, lane);
    f.b0 = load_b_frag(Wih, Kx, n0,       k, lane);
    f.b1 = load_b_frag(Wih, Kx, 128 + n0, k, lane);
    f.b2 = load_b_frag(Wih, Kx, 256 + n0, k, lane);
    f.b3 = load_b_frag(Wih, Kx, 384 + n0, k, lane);
  } else {
    int k = (kt - KxT) * 32;
    f.a  = load_a_frag(hA, 3 * HH, k, lane);
    f.b0 = load_b_frag(Whh, HH, n0,       k, lane);
    f.b1 = load_b_frag(Whh, HH, 128 + n0, k, lane);
    f.b2 = load_b_frag(Whh, HH, 256 + n0, k, lane);
    f.b3 = load_b_frag(Whh, HH, 384 + n0, k, lane);
  }
  return f;
}

// One LSTM cell: gates = x@Wih^T + h@Whh^T + (bih+bhh); wave w owns gate
// columns [16w,16w+16) for each of i/f/g/o. Double-buffered k-tile pipeline.
__device__ __forceinline__ void lstm_cell_stage(
    const _Float16* sX, int Kx, _Float16* sH16,
    const _Float16* Wih, const _Float16* Whh,
    const float* bih, const float* bhh,
    float* sC32, _Float16* sCstar, int cell, int lane, int wave) {
  v8f ai = {}, af = {}, ag = {}, ao = {};
  const int n0  = 16 * wave;
  const int KxT = Kx / 32;
  const int KT  = KxT + HH / 32;
  const _Float16* hA = sH16 + cell * HH;

  Frag5 cur = load_cell_frags(0, KxT, sX, Kx, hA, Wih, Whh, n0, lane);
#pragma unroll
  for (int kt = 1; kt < KT; ++kt) {
    Frag5 nxt = load_cell_frags(kt, KxT, sX, Kx, hA, Wih, Whh, n0, lane);
    ai = wmma32(ai, cur.a, cur.b0);
    af = wmma32(af, cur.a, cur.b1);
    ag = wmma32(ag, cur.a, cur.b2);
    ao = wmma32(ao, cur.a, cur.b3);
    cur = nxt;
  }
  ai = wmma32(ai, cur.a, cur.b0);
  af = wmma32(af, cur.a, cur.b1);
  ag = wmma32(ag, cur.a, cur.b2);
  ao = wmma32(ao, cur.a, cur.b3);

  __syncthreads();  // all waves done reading sH16[cell] before we overwrite it
  const int n  = n0 + (lane & 15);
  const int m0 = (lane >> 4) * 8;
  const float bi = bih[n]       + bhh[n];
  const float bf = bih[128 + n] + bhh[128 + n];
  const float bg = bih[256 + n] + bhh[256 + n];
  const float bo = bih[384 + n] + bhh[384 + n];
#pragma unroll
  for (int v = 0; v < 8; ++v) {
    int m = m0 + v;
    float gi = sigf(ai[v] + bi);
    float gf = sigf(af[v] + bf);
    float gg = tanhf(ag[v] + bg);
    float go = sigf(ao[v] + bo);
    float cp = sC32[m * 384 + cell * HH + n];
    float c2 = gf * cp + gi * gg;
    float h  = go * tanhf(c2);
    sC32[m * 384 + cell * HH + n] = c2;
    sCstar[m * 768 + 384 + cell * HH + n] = (_Float16)c2;   // new_cs half
    sH16[m * 384 + cell * HH + n] = (_Float16)h;
  }
}

struct Params {
  const float *texts, *audios, *videos;
  const float *bih_l, *bhh_l, *bih_a, *bhh_a, *bih_v, *bhh_v;
  const float *att1_b1, *att1_b2, *att2_b1, *att2_b2;
  const float *g1_b1, *g1_b2, *g2_b1, *g2_b2;
  const float *out_b1, *out_b2;
  const float *fc1_w, *fc1_b, *fc2_w, *fc2_b;
  const _Float16 *Wih_l, *Whh_l, *Wih_a, *Whh_a, *Wih_v, *Whh_v;
  const _Float16 *a1w1, *a1w2, *a2w1, *a2w2;
  const _Float16 *g1w1, *g1w2, *g2w1, *g2w2;
  const _Float16 *ow1, *ow2;
  float* out;
};

__global__ void cvt_f32_to_f16(const float* __restrict__ s, _Float16* __restrict__ d, int n) {
  int i = blockIdx.x * 256 + threadIdx.x;
  if (i < n) d[i] = (_Float16)s[i];
}

__global__ __launch_bounds__(256) void mfn_fused(Params p) {
  extern __shared__ __align__(16) char smem[];
  // fp32 region
  float* sC32 = (float*)smem;                    // [16][384]  c_l|c_a|c_v
  float* sMem = sC32 + 16 * 384;                 // [16][128]  mem (later sFeat)
  float* redA = sMem + 16 * 128;                 // [16][16]
  float* redB = redA + 256;                      // [16][16]
  // f16 region
  _Float16* sX     = (_Float16*)(redB + 256);    // [16][768]  x tile / last_hs
  _Float16* sH16   = sX + 16 * 768;              // [16][384]  h mirrors
  _Float16* sCstar = sH16 + 16 * 384;            // [16][768]  cStar f16
  _Float16* sLog   = sCstar + 16 * 768;          // [16][768]  logits / exp
  _Float16* sBoth  = sLog + 16 * 768;            // [16][896]  attended | mem16
  _Float16* sHid   = sBoth + 16 * 896;           // [16][128]  hidden f16

  const int tid  = threadIdx.x;
  const int lane = tid & 31;
  const int wave = tid >> 5;
  const int rowbase = blockIdx.x * 16;
  const int n0 = 16 * wave;

  // zero-init state
  for (int i = tid; i < 16 * 384; i += 256) { sC32[i] = 0.f; sH16[i] = (_Float16)0.f; }
  for (int i = tid; i < 16 * 128; i += 256) {
    sMem[i] = 0.f;
    sBoth[(i >> 7) * 896 + 768 + (i & 127)] = (_Float16)0.f;
  }
  __syncthreads();

  for (int t = 0; t < TT; ++t) {
    // prev_cs -> cStar[:, :384]
    for (int i = tid; i < 16 * 384; i += 256)
      sCstar[(i / 384) * 768 + (i % 384)] = (_Float16)sC32[i];
    load_x_tile(sX, p.texts, rowbase, t, TDI, tid);
    __syncthreads();
    lstm_cell_stage(sX, TDI, sH16, p.Wih_l, p.Whh_l, p.bih_l, p.bhh_l, sC32, sCstar, 0, lane, wave);
    __syncthreads();
    load_x_tile(sX, p.audios, rowbase, t, ADI, tid);
    __syncthreads();
    lstm_cell_stage(sX, ADI, sH16, p.Wih_a, p.Whh_a, p.bih_a, p.bhh_a, sC32, sCstar, 1, lane, wave);
    __syncthreads();
    load_x_tile(sX, p.videos, rowbase, t, VDI, tid);
    __syncthreads();
    lstm_cell_stage(sX, VDI, sH16, p.Wih_v, p.Whh_v, p.bih_v, p.bhh_v, sC32, sCstar, 2, lane, wave);
    __syncthreads();

    // att1 hidden: relu(cStar @ a1w1^T + b)   [16,128]
    { v8f acc = {};
      gemm_pipe(acc, sCstar, 768, 768, p.a1w1, 768, n0, lane);
      epilogue_f16<1>(sHid, 128, n0, acc, p.att1_b1, lane); }
    __syncthreads();
    // att logits: hid @ a1w2^T + b  [16,768]; A frags hoisted across 6 tiles
    {
      v16h A0 = load_a_frag(sHid, 128, 0,  lane);
      v16h A1 = load_a_frag(sHid, 128, 32, lane);
      v16h A2 = load_a_frag(sHid, 128, 64, lane);
      v16h A3 = load_a_frag(sHid, 128, 96, lane);
#pragma unroll
      for (int j = 0; j < 6; ++j) {
        int nt = 16 * (wave * 6 + j);
        v16h b0 = load_b_frag(p.a1w2, 128, nt, 0,  lane);
        v16h b1 = load_b_frag(p.a1w2, 128, nt, 32, lane);
        v16h b2 = load_b_frag(p.a1w2, 128, nt, 64, lane);
        v16h b3 = load_b_frag(p.a1w2, 128, nt, 96, lane);
        v8f acc = {};
        acc = wmma32(acc, A0, b0);
        acc = wmma32(acc, A1, b1);
        acc = wmma32(acc, A2, b2);
        acc = wmma32(acc, A3, b3);
        epilogue_f16<0>(sLog, 768, nt, acc, p.att1_b2, lane);
      }
    }
    __syncthreads();
    // softmax over 768 cols per row; attended = att * cStar -> sBoth[:, :768]
    {
      int r = tid >> 4, g = tid & 15;
      float lmax = -1e30f;
      for (int j = 0; j < 48; ++j)
        lmax = fmaxf(lmax, (float)sLog[r * 768 + g * 48 + j]);
      redA[r * 16 + g] = lmax;
      __syncthreads();
      float rmax = -1e30f;
      for (int q = 0; q < 16; ++q) rmax = fmaxf(rmax, redA[r * 16 + q]);
      float lsum = 0.f;
      for (int j = 0; j < 48; ++j) {
        int c = g * 48 + j;
        float e = __expf((float)sLog[r * 768 + c] - rmax);
        sLog[r * 768 + c] = (_Float16)e;   // e in (0,1], f16-safe
        lsum += e;
      }
      redB[r * 16 + g] = lsum;
      __syncthreads();
      float rsum = 0.f;
      for (int q = 0; q < 16; ++q) rsum += redB[r * 16 + q];
      float inv = 1.f / rsum;
      for (int j = 0; j < 48; ++j) {
        int c = g * 48 + j;
        float attd = (float)sLog[r * 768 + c] * inv * (float)sCstar[r * 768 + c];
        sBoth[r * 896 + c] = (_Float16)attd;
      }
    }
    __syncthreads();
    // att2 hidden: relu(attended @ a2w1^T + b)
    { v8f acc = {};
      gemm_pipe(acc, sBoth, 896, 768, p.a2w1, 768, n0, lane);
      epilogue_f16<1>(sHid, 128, n0, acc, p.att2_b1, lane); }
    __syncthreads();
    // cHat = tanh(hid @ a2w2^T + b) -- kept in registers (wave owns its cols)
    float ch[8];
    { v8f acc = gemm_k128(sHid, 128, p.a2w2, 128, n0, lane);
      float b = p.att2_b2[n0 + (lane & 15)];
#pragma unroll
      for (int v = 0; v < 8; ++v) ch[v] = tanhf(acc[v] + b); }
    __syncthreads();
    // g1 hidden: relu(both @ g1w1^T + b), K=896 includes mem cols
    { v8f acc = {};
      gemm_pipe(acc, sBoth, 896, 896, p.g1w1, 896, n0, lane);
      epilogue_f16<1>(sHid, 128, n0, acc, p.g1_b1, lane); }
    __syncthreads();
    // g1 = sigmoid(hid @ g1w2^T + b) -- registers
    float gv1[8];
    { v8f acc = gemm_k128(sHid, 128, p.g1w2, 128, n0, lane);
      float b = p.g1_b2[n0 + (lane & 15)];
#pragma unroll
      for (int v = 0; v < 8; ++v) gv1[v] = sigf(acc[v] + b); }
    __syncthreads();
    // g2 hidden
    { v8f acc = {};
      gemm_pipe(acc, sBoth, 896, 896, p.g2w1, 896, n0, lane);
      epilogue_f16<1>(sHid, 128, n0, acc, p.g2_b1, lane); }
    __syncthreads();
    // g2 + memory update: mem = g1*mem + g2*cHat
    { v8f acc = gemm_k128(sHid, 128, p.g2w2, 128, n0, lane);
      int n = n0 + (lane & 15);
      float b = p.g2_b2[n];
      int m0 = (lane >> 4) * 8;
#pragma unroll
      for (int v = 0; v < 8; ++v) {
        float g2v = sigf(acc[v] + b);
        int m = m0 + v;
        float mv = sMem[m * 128 + n];
        mv = gv1[v] * mv + g2v * ch[v];
        sMem[m * 128 + n] = mv;
        sBoth[m * 896 + 768 + n] = (_Float16)mv;  // mem f16 mirror for next step
      }
    }
    __syncthreads();
  }

  // last_hs = [h_l|h_a|h_v|mem] f16 into sX [16][512]
  for (int i = tid; i < 16 * 512; i += 256) {
    int r = i >> 9, c = i & 511;
    sX[r * 512 + c] = (c < 384) ? sH16[r * 384 + c]
                                : sBoth[r * 896 + 768 + (c - 384)];
  }
  __syncthreads();
  // relu(last_hs @ out_w1^T + b)  [16,128]
  { v8f acc = {};
    gemm_pipe(acc, sX, 512, 512, p.ow1, 512, n0, lane);
    epilogue_f16<1>(sHid, 128, n0, acc, p.out_b1, lane); }
  __syncthreads();
  // features = hid @ out_w2^T + b  [16,64]  (waves 0..3)
  float* sFeat = sMem;  // reuse as [16][64] fp32
  if (wave < 4) {
    v8f acc = gemm_k128(sHid, 128, p.ow2, 128, n0, lane);
    int n = n0 + (lane & 15);
    float b = p.out_b2[n];
    int m0 = (lane >> 4) * 8;
#pragma unroll
    for (int v = 0; v < 8; ++v) {
      float f = acc[v] + b;
      int m = m0 + v;
      sFeat[m * 64 + n] = f;
      p.out[(size_t)(rowbase + m) * 64 + n] = f;
    }
  }
  __syncthreads();
  // emos [256,6] and vals [256,1] heads (tiny, scalar)
  if (tid < 16 * 7) {
    int r = tid / 7, j = tid - r * 7;
    if (j < 6) {
      float s = p.fc1_b[j];
      for (int q = 0; q < 64; ++q) s += sFeat[r * 64 + q] * p.fc1_w[j * 64 + q];
      p.out[16384 + (size_t)(rowbase + r) * 6 + j] = s;
    } else {
      float s = p.fc2_b[0];
      for (int q = 0; q < 64; ++q) s += sFeat[r * 64 + q] * p.fc2_w[q];
      p.out[16384 + 1536 + (rowbase + r)] = s;
    }
  }
  if (blockIdx.x == 0 && tid == 0) p.out[18176] = 0.f;  // trailing scalar 0
}

extern "C" void kernel_launch(void* const* d_in, const int* in_sizes, int n_in,
                              void* d_out, int out_size, void* d_ws, size_t ws_size,
                              hipStream_t stream) {
  (void)in_sizes; (void)n_in; (void)out_size; (void)ws_size;
  _Float16* ws = (_Float16*)d_ws;
  size_t off = 0;
  auto take = [&](size_t n) { _Float16* q = ws + off; off += n; return q; };
  _Float16* wWih_l = take(512 * 768);
  _Float16* wWhh_l = take(512 * 128);
  _Float16* wWih_a = take(512 * 512);
  _Float16* wWhh_a = take(512 * 128);
  _Float16* wWih_v = take(512 * 512);
  _Float16* wWhh_v = take(512 * 128);
  _Float16* wA1w1  = take(128 * 768);
  _Float16* wA1w2  = take(768 * 128);
  _Float16* wA2w1  = take(128 * 768);
  _Float16* wA2w2  = take(128 * 128);
  _Float16* wG1w1  = take(128 * 896);
  _Float16* wG1w2  = take(128 * 128);
  _Float16* wG2w1  = take(128 * 896);
  _Float16* wG2w2  = take(128 * 128);
  _Float16* wOw1   = take(128 * 512);
  _Float16* wOw2   = take(64 * 128);

  struct Job { const float* s; _Float16* d; int n; };
  Job jobs[16] = {
    {(const float*)d_in[3],  wWih_l, 512 * 768},
    {(const float*)d_in[4],  wWhh_l, 512 * 128},
    {(const float*)d_in[7],  wWih_a, 512 * 512},
    {(const float*)d_in[8],  wWhh_a, 512 * 128},
    {(const float*)d_in[11], wWih_v, 512 * 512},
    {(const float*)d_in[12], wWhh_v, 512 * 128},
    {(const float*)d_in[15], wA1w1,  128 * 768},
    {(const float*)d_in[17], wA1w2,  768 * 128},
    {(const float*)d_in[19], wA2w1,  128 * 768},
    {(const float*)d_in[21], wA2w2,  128 * 128},
    {(const float*)d_in[23], wG1w1,  128 * 896},
    {(const float*)d_in[25], wG1w2,  128 * 128},
    {(const float*)d_in[27], wG2w1,  128 * 896},
    {(const float*)d_in[29], wG2w2,  128 * 128},
    {(const float*)d_in[31], wOw1,   128 * 512},
    {(const float*)d_in[33], wOw2,    64 * 128},
  };
  for (int i = 0; i < 16; ++i)
    cvt_f32_to_f16<<<(jobs[i].n + 255) / 256, 256, 0, stream>>>(jobs[i].s, jobs[i].d, jobs[i].n);

  Params p;
  p.texts  = (const float*)d_in[0];
  p.audios = (const float*)d_in[1];
  p.videos = (const float*)d_in[2];
  p.bih_l = (const float*)d_in[5];  p.bhh_l = (const float*)d_in[6];
  p.bih_a = (const float*)d_in[9];  p.bhh_a = (const float*)d_in[10];
  p.bih_v = (const float*)d_in[13]; p.bhh_v = (const float*)d_in[14];
  p.att1_b1 = (const float*)d_in[16]; p.att1_b2 = (const float*)d_in[18];
  p.att2_b1 = (const float*)d_in[20]; p.att2_b2 = (const float*)d_in[22];
  p.g1_b1 = (const float*)d_in[24]; p.g1_b2 = (const float*)d_in[26];
  p.g2_b1 = (const float*)d_in[28]; p.g2_b2 = (const float*)d_in[30];
  p.out_b1 = (const float*)d_in[32]; p.out_b2 = (const float*)d_in[34];
  p.fc1_w = (const float*)d_in[35]; p.fc1_b = (const float*)d_in[36];
  p.fc2_w = (const float*)d_in[37]; p.fc2_b = (const float*)d_in[38];
  p.Wih_l = wWih_l; p.Whh_l = wWhh_l;
  p.Wih_a = wWih_a; p.Whh_a = wWhh_a;
  p.Wih_v = wWih_v; p.Whh_v = wWhh_v;
  p.a1w1 = wA1w1; p.a1w2 = wA1w2; p.a2w1 = wA2w1; p.a2w2 = wA2w2;
  p.g1w1 = wG1w1; p.g1w2 = wG1w2; p.g2w1 = wG2w1; p.g2w2 = wG2w2;
  p.ow1 = wOw1; p.ow2 = wOw2;
  p.out = (float*)d_out;

  mfn_fused<<<dim3(BB / 16), dim3(256), SMEM_BYTES, stream>>>(p);
}
